// GNNBranch_86174223827123
// MI455X (gfx1250) — matrix operand order repaired and checked by reading, not compiled, not run on hardware
//
#include <hip/hip_runtime.h>

typedef __attribute__((ext_vector_type(16))) __bf16 v16bf;
typedef __attribute__((ext_vector_type(8)))  float  v8f;

#define N_NODES   8192
#define NGRAPH    16
#define PER_G     512
#define NEDGE     131072
#define NEDGE_SL  (NEDGE + N_NODES)
#define PDIM_IN   1310
#define KP_P      1344
#define LDIM_IN   36
#define KP_L      64
#define DMODEL    256
#define NHEAD     4
#define HDIM      1024
#define ADIM_P    128
#define NLAYER    3
#define NEGV      (-1.0e30f)

// ---------- helpers ----------
__device__ __forceinline__ unsigned short f2bf(float f) {
  unsigned u = __float_as_uint(f);
  u += 0x7FFFu + ((u >> 16) & 1u);          // round-to-nearest-even
  return (unsigned short)(u >> 16);
}

// order-preserving float<->uint map for atomicMax on floats
__device__ __forceinline__ unsigned fkey(float f) {
  unsigned b = __float_as_uint(f);
  return (b & 0x80000000u) ? ~b : (b | 0x80000000u);
}
__device__ __forceinline__ float funkey(unsigned k) {
  unsigned b = (k & 0x80000000u) ? (k ^ 0x80000000u) : ~k;
  return __uint_as_float(b);
}

__device__ __forceinline__ float block_red_sum(float v, float* sbuf) {
  int t = threadIdx.x;
  sbuf[t] = v; __syncthreads();
  for (int s = blockDim.x >> 1; s > 0; s >>= 1) {
    if (t < s) sbuf[t] += sbuf[t + s];
    __syncthreads();
  }
  float r = sbuf[0]; __syncthreads();
  return r;
}
__device__ __forceinline__ float block_red_max(float v, float* sbuf) {
  int t = threadIdx.x;
  sbuf[t] = v; __syncthreads();
  for (int s = blockDim.x >> 1; s > 0; s >>= 1) {
    if (t < s) sbuf[t] = fmaxf(sbuf[t], sbuf[t + s]);
    __syncthreads();
  }
  float r = sbuf[0]; __syncthreads();
  return r;
}

// ---------- conversion kernels ----------
// take cols [0,Kin) of each row of X[N, ldx], emit bf16 padded to Kp cols
__global__ void conv_rows_kernel(const float* __restrict__ X, unsigned short* __restrict__ XB,
                                 int ldx, int Kin, int Kp, long total) {
  long i = (long)blockIdx.x * blockDim.x + threadIdx.x;
  if (i >= total) return;
  int n = (int)(i / Kp), k = (int)(i % Kp);
  XB[i] = (k < Kin) ? f2bf(X[(size_t)n * ldx + k]) : (unsigned short)0;
}

// W[K, Nn] row-major  ->  WT[Nn, Kp] bf16 (transposed, K padded)
__global__ void conv_transpose_kernel(const float* __restrict__ W, unsigned short* __restrict__ WT,
                                      int K, int Nn, int Kp, long total) {
  long i = (long)blockIdx.x * blockDim.x + threadIdx.x;
  if (i >= total) return;
  int n = (int)(i / Kp), k = (int)(i % Kp);
  WT[i] = (k < K) ? f2bf(W[(size_t)k * Nn + n]) : (unsigned short)0;
}

__global__ void zero_u32_kernel(unsigned* __restrict__ p, long n) {
  long i = (long)blockIdx.x * blockDim.x + threadIdx.x;
  if (i < n) p[i] = 0u;
}

// ---------- WMMA GEMM:  C[M,N] = A_bf16[M,K] * BT_bf16[N,K]^T (+bias[N]) ----------
union FragU { v16bf v; uint4 q[2]; };

__global__ void wmma_gemm_bias_kernel(const unsigned short* __restrict__ A,
                                      const unsigned short* __restrict__ BT,
                                      const float* __restrict__ bias,
                                      float* __restrict__ C,
                                      int M, int N, int K) {
  const int lane = threadIdx.x & 31;
  const int wave = threadIdx.x >> 5;          // 0..3
  const int m0 = blockIdx.y * 64 + wave * 16;
  const int n0 = blockIdx.x * 64;
  if (m0 >= M) return;
  const int l15 = lane & 15;
  const int hi  = lane >> 4;
  const int row = m0 + l15;
  const int ka  = hi << 3;   // A: upper half-wave holds K+8..
  const int kb  = hi << 3;   // B: upper half-wave holds K+8..

  v8f c0 = {}, c1 = {}, c2 = {}, c3 = {};
  const unsigned short* arow = A + (size_t)row * K;
  const unsigned short* b0p = BT + (size_t)(n0 +  0 + l15) * K + kb;
  const unsigned short* b1p = BT + (size_t)(n0 + 16 + l15) * K + kb;
  const unsigned short* b2p = BT + (size_t)(n0 + 32 + l15) * K + kb;
  const unsigned short* b3p = BT + (size_t)(n0 + 48 + l15) * K + kb;

  for (int k0 = 0; k0 < K; k0 += 32) {
    FragU a, b0, b1, b2, b3;
    a.q[0]  = *(const uint4*)(arow + k0 + ka);
    a.q[1]  = *(const uint4*)(arow + k0 + ka + 16);
    b0.q[0] = *(const uint4*)(b0p + k0);  b0.q[1] = *(const uint4*)(b0p + k0 + 16);
    b1.q[0] = *(const uint4*)(b1p + k0);  b1.q[1] = *(const uint4*)(b1p + k0 + 16);
    b2.q[0] = *(const uint4*)(b2p + k0);  b2.q[1] = *(const uint4*)(b2p + k0 + 16);
    b3.q[0] = *(const uint4*)(b3p + k0);  b3.q[1] = *(const uint4*)(b3p + k0 + 16);
    c0 = __builtin_amdgcn_wmma_f32_16x16x32_bf16(false, a.v, false, b0.v, (short)0, c0, false, false);
    c1 = __builtin_amdgcn_wmma_f32_16x16x32_bf16(false, a.v, false, b1.v, (short)0, c1, false, false);
    c2 = __builtin_amdgcn_wmma_f32_16x16x32_bf16(false, a.v, false, b2.v, (short)0, c2, false, false);
    c3 = __builtin_amdgcn_wmma_f32_16x16x32_bf16(false, a.v, false, b3.v, (short)0, c3, false, false);
  }

  float bias0 = 0.f, bias1 = 0.f, bias2 = 0.f, bias3 = 0.f;
  if (bias) {
    bias0 = bias[n0 + l15];       bias1 = bias[n0 + 16 + l15];
    bias2 = bias[n0 + 32 + l15];  bias3 = bias[n0 + 48 + l15];
  }
  const int rbase = m0 + (hi << 3);
#pragma unroll
  for (int r = 0; r < 8; ++r) {
    float* crow = C + (size_t)(rbase + r) * N + n0 + l15;
    crow[0]  = c0[r] + bias0;
    crow[16] = c1[r] + bias1;
    crow[32] = c2[r] + bias2;
    crow[48] = c3[r] + bias3;
  }
}

// ---------- h = relu(LN(sel)) + node_emb[nt] ----------
__global__ void fuse_h_kernel(const float* __restrict__ prot, const float* __restrict__ lig,
                              const int* __restrict__ node_type,
                              const float* __restrict__ gp, const float* __restrict__ betap,
                              const float* __restrict__ gl, const float* __restrict__ betal,
                              const float* __restrict__ node_emb,
                              float* __restrict__ h, unsigned short* __restrict__ hb) {
  __shared__ float sbuf[DMODEL];
  const int n = blockIdx.x, d = threadIdx.x;
  const int nt = node_type[n];
  const float* src = (nt == 0 ? prot : lig) + (size_t)n * DMODEL;
  float v = src[d];
  float mean = block_red_sum(v, sbuf) * (1.0f / DMODEL);
  float dv = v - mean;
  float var = block_red_sum(dv * dv, sbuf) * (1.0f / DMODEL);
  float g  = (nt == 0) ? gp[d] : gl[d];
  float be = (nt == 0) ? betap[d] : betal[d];
  float y = dv * rsqrtf(var + 1e-5f) * g + be;
  y = fmaxf(y, 0.f) + node_emb[nt * DMODEL + d];
  h [(size_t)n * DMODEL + d] = y;
  hb[(size_t)n * DMODEL + d] = f2bf(y);
}

// ---------- a_s / a_d coefficients ----------
__global__ void coef_kernel(const float* __restrict__ xh,
                            const float* __restrict__ asrc, const float* __restrict__ adst,
                            float* __restrict__ a_s, float* __restrict__ a_d) {
  __shared__ float sbuf[DMODEL];
  const int n = blockIdx.x, d = threadIdx.x;
  for (int hh = 0; hh < NHEAD; ++hh) {
    float xv = xh[(size_t)n * HDIM + hh * DMODEL + d];
    float ss = block_red_sum(xv * asrc[hh * DMODEL + d], sbuf);
    float sd = block_red_sum(xv * adst[hh * DMODEL + d], sbuf);
    if (d == 0) { a_s[n * NHEAD + hh] = ss; a_d[n * NHEAD + hh] = sd; }
  }
}

// ---------- edge kernels ----------
__device__ __forceinline__ void edge_sd(const int* ei, int e, int& src, int& dst) {
  if (e < NEDGE) { src = ei[e]; dst = ei[NEDGE + e]; }
  else           { src = dst = e - NEDGE; }          // self-loops appended
}

__global__ void edge_max_kernel(const int* __restrict__ ei,
                                const float* __restrict__ a_s, const float* __restrict__ a_d,
                                unsigned* __restrict__ mseg) {
  int e = blockIdx.x * blockDim.x + threadIdx.x;
  if (e >= NEDGE_SL) return;
  int src, dst; edge_sd(ei, e, src, dst);
  for (int hh = 0; hh < NHEAD; ++hh) {
    float v = a_s[src * NHEAD + hh] + a_d[dst * NHEAD + hh];
    float lr = v > 0.f ? v : 0.2f * v;
    atomicMax(&mseg[dst * NHEAD + hh], fkey(lr));
  }
}

__global__ void edge_expsum_kernel(const int* __restrict__ ei,
                                   const float* __restrict__ a_s, const float* __restrict__ a_d,
                                   const unsigned* __restrict__ mseg,
                                   float* __restrict__ ssum, float* __restrict__ ealpha) {
  int e = blockIdx.x * blockDim.x + threadIdx.x;
  if (e >= NEDGE_SL) return;
  int src, dst; edge_sd(ei, e, src, dst);
  for (int hh = 0; hh < NHEAD; ++hh) {
    float v = a_s[src * NHEAD + hh] + a_d[dst * NHEAD + hh];
    float lr = v > 0.f ? v : 0.2f * v;
    float ee = expf(lr - funkey(mseg[dst * NHEAD + hh]));
    ealpha[(size_t)e * NHEAD + hh] = ee;
    atomicAdd(&ssum[dst * NHEAD + hh], ee);
  }
}

__global__ void edge_aggregate_kernel(const int* __restrict__ ei,
                                      const float* __restrict__ xh,
                                      const float* __restrict__ ealpha,
                                      const float* __restrict__ ssum,
                                      float* __restrict__ xn) {
  const int e = blockIdx.x, d = threadIdx.x;   // 256 threads = feature dim
  int src, dst; edge_sd(ei, e, src, dst);
  float acc = 0.f;
  for (int hh = 0; hh < NHEAD; ++hh) {
    float w = ealpha[(size_t)e * NHEAD + hh] / ssum[dst * NHEAD + hh];
    acc += xh[(size_t)src * HDIM + hh * DMODEL + d] * w;
  }
  atomicAdd(&xn[(size_t)dst * DMODEL + d], acc * (1.0f / NHEAD));  // head-mean folded in
}

// ---------- xn + gat_b (+residual) -> LN -> ReLU -> h, hb ----------
__global__ void post_layer_kernel(const float* __restrict__ xn, const float* __restrict__ gat_b,
                                  const float* __restrict__ ln_g, const float* __restrict__ ln_b,
                                  float* __restrict__ h, unsigned short* __restrict__ hb,
                                  int use_residual) {
  __shared__ float sbuf[DMODEL];
  const int n = blockIdx.x, d = threadIdx.x;
  float v = xn[(size_t)n * DMODEL + d] + gat_b[d];
  if (use_residual) v += h[(size_t)n * DMODEL + d];
  float mean = block_red_sum(v, sbuf) * (1.0f / DMODEL);
  float dv = v - mean;
  float var = block_red_sum(dv * dv, sbuf) * (1.0f / DMODEL);
  float y = dv * rsqrtf(var + 1e-5f) * ln_g[d] + ln_b[d];
  y = fmaxf(y, 0.f);
  h [(size_t)n * DMODEL + d] = y;
  hb[(size_t)n * DMODEL + d] = f2bf(y);
}

// ---------- scores = tanh(t1) @ W2 + b2 ----------
__global__ void scores_kernel(const float* __restrict__ t1, const float* __restrict__ W2,
                              const float* __restrict__ b2, float* __restrict__ scores) {
  __shared__ float sbuf[ADIM_P];
  const int n = blockIdx.x, t = threadIdx.x;    // 128 threads
  float tv = tanhf(t1[(size_t)n * ADIM_P + t]);
  for (int hh = 0; hh < NHEAD; ++hh) {
    float s = block_red_sum(tv * W2[t * NHEAD + hh], sbuf);
    if (t == 0) scores[n * NHEAD + hh] = s + b2[hh];
  }
}

// ---------- per-graph attention pooling + masked mean ----------
__global__ void pool_kernel(const float* __restrict__ h, const float* __restrict__ scores,
                            const int* __restrict__ node_type, float* __restrict__ out) {
  const int g = blockIdx.x, t = threadIdx.x;    // 256 threads
  const int base = g * PER_G;
  __shared__ float red[DMODEL];
  __shared__ float wsum[PER_G];
  __shared__ int   snt[PER_G];
  __shared__ float s_mm[NHEAD], s_mu[NHEAD], s_es[NHEAD], s_eus[NHEAD];
  __shared__ int s_hp;

  float pmax[NHEAD], umax[NHEAD];
  for (int hh = 0; hh < NHEAD; ++hh) { pmax[hh] = -3.0e38f; umax[hh] = -3.0e38f; }
  float pcnt = 0.f;
  for (int n = t; n < PER_G; n += DMODEL) {
    int nt = node_type[base + n];
    snt[n] = nt;
    if (nt == 0) pcnt += 1.f;
    for (int hh = 0; hh < NHEAD; ++hh) {
      float s = scores[(size_t)(base + n) * NHEAD + hh];
      umax[hh] = fmaxf(umax[hh], s);
      pmax[hh] = fmaxf(pmax[hh], nt == 0 ? s : NEGV);
    }
  }
  for (int hh = 0; hh < NHEAD; ++hh) {
    float r = block_red_max(pmax[hh], red); if (t == 0) s_mm[hh] = r;
    r = block_red_max(umax[hh], red);       if (t == 0) s_mu[hh] = r;
  }
  float cp = block_red_sum(pcnt, red);
  if (t == 0) s_hp = (cp > 0.f) ? 1 : 0;
  __syncthreads();

  float es[NHEAD] = {0, 0, 0, 0}, eus[NHEAD] = {0, 0, 0, 0};
  for (int n = t; n < PER_G; n += DMODEL) {
    int nt = snt[n];
    for (int hh = 0; hh < NHEAD; ++hh) {
      float s = scores[(size_t)(base + n) * NHEAD + hh];
      float masked = (nt == 0) ? s : NEGV;
      es [hh] += expf(masked - s_mm[hh]);
      eus[hh] += expf(s - s_mu[hh]);
    }
  }
  for (int hh = 0; hh < NHEAD; ++hh) {
    float r = block_red_sum(es[hh], red);  if (t == 0) s_es[hh]  = r;
    r = block_red_sum(eus[hh], red);       if (t == 0) s_eus[hh] = r;
  }
  __syncthreads();

  for (int n = t; n < PER_G; n += DMODEL) {
    int nt = snt[n];
    float ws = 0.f;
    for (int hh = 0; hh < NHEAD; ++hh) {
      float s = scores[(size_t)(base + n) * NHEAD + hh];
      float w;
      if (s_hp) {
        float masked = (nt == 0) ? s : NEGV;
        w = expf(masked - s_mm[hh]) / fmaxf(s_es[hh], 1e-30f);
      } else {
        w = expf(s - s_mu[hh]) / fmaxf(s_eus[hh], 1e-30f);
      }
      ws += w;
    }
    wsum[n] = ws * (1.0f / NHEAD);
  }
  __syncthreads();

  float pacc = 0.f, gacc = 0.f;
  for (int n = 0; n < PER_G; ++n) {
    float hv = h[(size_t)(base + n) * DMODEL + t];
    pacc += hv * wsum[n];
    if (snt[n] == 0) gacc += hv;
  }
  out[(size_t)g * (2 * DMODEL) + t]          = pacc;
  out[(size_t)g * (2 * DMODEL) + DMODEL + t] = gacc * (1.0f / PER_G);
}

// ---------- launcher ----------
extern "C" void kernel_launch(void* const* d_in, const int* in_sizes, int n_in,
                              void* d_out, int out_size, void* d_ws, size_t ws_size,
                              hipStream_t stream) {
  (void)in_sizes; (void)n_in; (void)out_size; (void)ws_size;
  const float* x        = (const float*)d_in[0];
  const int*   ei       = (const int*)d_in[1];
  const int*   node_type= (const int*)d_in[3];
  const float* Wp       = (const float*)d_in[4];
  const float* bp       = (const float*)d_in[5];
  const float* gp       = (const float*)d_in[6];
  const float* betap    = (const float*)d_in[7];
  const float* Wl       = (const float*)d_in[8];
  const float* bl       = (const float*)d_in[9];
  const float* gl       = (const float*)d_in[10];
  const float* betal    = (const float*)d_in[11];
  const float* node_emb = (const float*)d_in[12];
  const float* gat_W    = (const float*)d_in[13];
  const float* gat_asrc = (const float*)d_in[14];
  const float* gat_adst = (const float*)d_in[15];
  const float* gat_b    = (const float*)d_in[16];
  const float* ln_g     = (const float*)d_in[17];
  const float* ln_b     = (const float*)d_in[18];
  const float* W1       = (const float*)d_in[19];
  const float* b1       = (const float*)d_in[20];
  const float* W2       = (const float*)d_in[21];
  const float* b2       = (const float*)d_in[22];
  float* out = (float*)d_out;

  char* wsp = (char*)d_ws;
  auto alloc = [&](size_t bytes) -> char* {
    char* p = wsp; wsp += (bytes + 255) & ~(size_t)255; return p;
  };
  unsigned short* xb   = (unsigned short*)alloc((size_t)N_NODES * KP_P * 2);
  unsigned short* xlb  = (unsigned short*)alloc((size_t)N_NODES * KP_L * 2);
  unsigned short* WpT  = (unsigned short*)alloc((size_t)DMODEL * KP_P * 2);
  unsigned short* WlT  = (unsigned short*)alloc((size_t)DMODEL * KP_L * 2);
  unsigned short* WgT  = (unsigned short*)alloc((size_t)NLAYER * HDIM * DMODEL * 2);
  unsigned short* W1T  = (unsigned short*)alloc((size_t)ADIM_P * DMODEL * 2);
  float* prot  = (float*)alloc((size_t)N_NODES * DMODEL * 4);   // reused as xn
  float* lig   = (float*)alloc((size_t)N_NODES * DMODEL * 4);   // reused as t1
  float* h     = (float*)alloc((size_t)N_NODES * DMODEL * 4);
  unsigned short* hb = (unsigned short*)alloc((size_t)N_NODES * DMODEL * 2);
  float* xh    = (float*)alloc((size_t)N_NODES * HDIM * 4);
  float* a_s   = (float*)alloc((size_t)N_NODES * NHEAD * 4);
  float* a_d   = (float*)alloc((size_t)N_NODES * NHEAD * 4);
  unsigned* mseg = (unsigned*)alloc((size_t)N_NODES * NHEAD * 4);
  float* ssum  = (float*)alloc((size_t)N_NODES * NHEAD * 4);
  float* ealpha= (float*)alloc((size_t)NEDGE_SL * NHEAD * 4);
  float* xn = prot;
  float* t1 = lig;
  float* scores = a_s;   // a_s dead after last layer

  const int TB = 256;
  auto cdiv = [](long a, long b) { return (int)((a + b - 1) / b); };

  // 1) conversions
  {
    long tx = (long)N_NODES * KP_P;
    conv_rows_kernel<<<cdiv(tx, TB), TB, 0, stream>>>(x, xb, PDIM_IN, PDIM_IN, KP_P, tx);
    long tl = (long)N_NODES * KP_L;
    conv_rows_kernel<<<cdiv(tl, TB), TB, 0, stream>>>(x, xlb, PDIM_IN, LDIM_IN, KP_L, tl);
    long tp = (long)DMODEL * KP_P;
    conv_transpose_kernel<<<cdiv(tp, TB), TB, 0, stream>>>(Wp, WpT, PDIM_IN, DMODEL, KP_P, tp);
    long tw = (long)DMODEL * KP_L;
    conv_transpose_kernel<<<cdiv(tw, TB), TB, 0, stream>>>(Wl, WlT, LDIM_IN, DMODEL, KP_L, tw);
    long tg = (long)HDIM * DMODEL;
    for (int l = 0; l < NLAYER; ++l)
      conv_transpose_kernel<<<cdiv(tg, TB), TB, 0, stream>>>(
          gat_W + (size_t)l * DMODEL * HDIM, WgT + (size_t)l * HDIM * DMODEL,
          DMODEL, HDIM, DMODEL, tg);
    long t1n = (long)ADIM_P * DMODEL;
    conv_transpose_kernel<<<cdiv(t1n, TB), TB, 0, stream>>>(W1, W1T, DMODEL, ADIM_P, DMODEL, t1n);
  }

  // 2) input projections (WMMA)
  wmma_gemm_bias_kernel<<<dim3(DMODEL / 64, N_NODES / 64), 128, 0, stream>>>(
      xb, WpT, bp, prot, N_NODES, DMODEL, KP_P);
  wmma_gemm_bias_kernel<<<dim3(DMODEL / 64, N_NODES / 64), 128, 0, stream>>>(
      xlb, WlT, bl, lig, N_NODES, DMODEL, KP_L);
  fuse_h_kernel<<<N_NODES, DMODEL, 0, stream>>>(prot, lig, node_type, gp, betap, gl, betal,
                                                node_emb, h, hb);

  // 3) GAT layers
  for (int l = 0; l < NLAYER; ++l) {
    wmma_gemm_bias_kernel<<<dim3(HDIM / 64, N_NODES / 64), 128, 0, stream>>>(
        hb, WgT + (size_t)l * HDIM * DMODEL, (const float*)nullptr, xh, N_NODES, HDIM, DMODEL);
    coef_kernel<<<N_NODES, DMODEL, 0, stream>>>(xh, gat_asrc + (size_t)l * NHEAD * DMODEL,
                                                gat_adst + (size_t)l * NHEAD * DMODEL, a_s, a_d);
    zero_u32_kernel<<<cdiv((long)N_NODES * NHEAD, TB), TB, 0, stream>>>(mseg, (long)N_NODES * NHEAD);
    zero_u32_kernel<<<cdiv((long)N_NODES * NHEAD, TB), TB, 0, stream>>>((unsigned*)ssum, (long)N_NODES * NHEAD);
    edge_max_kernel<<<cdiv(NEDGE_SL, TB), TB, 0, stream>>>(ei, a_s, a_d, mseg);
    edge_expsum_kernel<<<cdiv(NEDGE_SL, TB), TB, 0, stream>>>(ei, a_s, a_d, mseg, ssum, ealpha);
    zero_u32_kernel<<<cdiv((long)N_NODES * DMODEL, TB), TB, 0, stream>>>((unsigned*)xn, (long)N_NODES * DMODEL);
    edge_aggregate_kernel<<<NEDGE_SL, DMODEL, 0, stream>>>(ei, xh, ealpha, ssum, xn);
    post_layer_kernel<<<N_NODES, DMODEL, 0, stream>>>(xn, gat_b + (size_t)l * DMODEL,
                                                      ln_g + (size_t)l * DMODEL,
                                                      ln_b + (size_t)l * DMODEL, h, hb, l > 0);
  }

  // 4) attention pooling
  wmma_gemm_bias_kernel<<<dim3(ADIM_P / 64, N_NODES / 64), 128, 0, stream>>>(
      hb, W1T, b1, t1, N_NODES, ADIM_P, DMODEL);
  scores_kernel<<<N_NODES, ADIM_P, 0, stream>>>(t1, W2, b2, scores);
  pool_kernel<<<NGRAPH, DMODEL, 0, stream>>>(h, scores, node_type, out);
}
